// SpikingConv3DLayer_fast_11381663334467
// MI455X (gfx1250) — matrix operand order repaired and checked by reading, not compiled, not run on hardware
//
#include <hip/hip_runtime.h>

// Problem constants (reference: B=4, CIN=32, T=50, H=W=64, COUT=64, K=3x3x3)
#define B_    4
#define CIN   32
#define T_    50
#define H_    64
#define W_    64
#define COUT  64
#define KTAPS 27
#define KCIN  (CIN * KTAPS)              // 864 reduction length
#define KCINP 872                        // padded weight row (shorts): 436 dwords -> 16 distinct banks
#define OUT_N (B_ * COUT * T_ * H_ * W_) // 52,428,800 spk_rec elements

// x-patch ring in LDS: 4 t-slices (double-buffer distance 2 for async overlap),
// each 3 h-rows x 66 w (halo) x cin padded 32->40 shorts (80 B: 16B-aligned pixel
// base, 20-bank pixel stride).
#define WPAD  66
#define CPAD  40
#define XSL   (3 * WPAD * CPAD)          // shorts per t-slot (7920)
#define NSLOT 4
#define WLDS_BYTES (COUT * KCINP * 2)    // 111,616
#define XLDS_BYTES (NSLOT * XSL * 2)     // 63,360

typedef __attribute__((ext_vector_type(16))) __bf16       v16bf;
typedef __attribute__((ext_vector_type(8)))  float        v8f;
typedef __attribute__((ext_vector_type(4)))  unsigned int v4u;

union Frag { v16bf bf; v4u q[2]; };

__device__ __forceinline__ unsigned short f2bf(float f) {
  unsigned int u = __builtin_bit_cast(unsigned int, f);
  u = (u + 0x7FFFu + ((u >> 16) & 1u)) >> 16;   // round-to-nearest-even
  return (unsigned short)u;
}

// CDNA5 async global->LDS 16B copy (ASYNCcnt-tracked; ISA §10.7/§15.18.3).
__device__ __forceinline__ void async_g2l_b128(const void* gptr, void* lptr) {
  unsigned int loff = (unsigned int)(unsigned long long)lptr; // LDS aperture: addr[31:0]
  asm volatile("global_load_async_to_lds_b128 %0, %1, off"
               :: "v"(loff), "v"(gptr) : "memory");
}

__device__ __forceinline__ void wait_async0() {
#if __has_builtin(__builtin_amdgcn_s_wait_asynccnt)
  __builtin_amdgcn_s_wait_asynccnt(0);
#else
  asm volatile("s_wait_asynccnt 0" ::: "memory");
#endif
}

// ---------------------------------------------------------------------------
// Kernel 0: per-cout norm=sum(w^2); fold scale=1/(norm+eps) into bf16 weights
// laid out [cout][tap][cin] (cin contiguous = WMMA K order). Zero counters.
// ---------------------------------------------------------------------------
__global__ void __launch_bounds__(256)
prep_kernel(const float* __restrict__ w, unsigned short* __restrict__ wb,
            float* __restrict__ cnt) {
  int tid = threadIdx.x;
  if (tid < COUT) {
    const float* wr = w + tid * KCIN;   // [cin][kd][kh][kw]
    float norm = 0.f;
    for (int i = 0; i < KCIN; ++i) { float v = wr[i]; norm += v * v; }
    float scale = 1.0f / (norm + 1e-8f);
    unsigned short* dst = wb + tid * KCIN;
    for (int cin = 0; cin < CIN; ++cin)
      for (int tap = 0; tap < KTAPS; ++tap)
        dst[tap * CIN + cin] = f2bf(wr[cin * KTAPS + tap] * scale);
  }
  if (tid >= 64 && tid < 64 + T_ + 2) cnt[tid - 64] = 0.f;
}

// ---------------------------------------------------------------------------
// Kernel 1: transpose+convert x: f32 [b][cin][t][h][w] -> bf16 [b][t][h][w][cin].
// cin-contiguous 64B runs enable async b128 staging in the conv kernel and
// halve x bytes there. One block per (b,t,h) row; LDS transpose keeps both
// global phases coalesced.
// ---------------------------------------------------------------------------
__global__ void __launch_bounds__(256)
xpose_kernel(const float* __restrict__ x, unsigned short* __restrict__ xb) {
  __shared__ float tl[W_][CIN + 1];     // +1 pad: conflict-free transpose
  const int idx = blockIdx.x;           // (b*T + t)*H + h
  const int h = idx & 63;
  const int t = (idx >> 6) % T_;
  const int b = idx / (T_ * H_);
  const int tid = threadIdx.x;

  const float* src = x + ((size_t)(b * CIN) * T_ + t) * (H_ * W_) + h * W_;
  for (int e = tid; e < CIN * W_; e += 256) {
    int wv = e & 63, cin = e >> 6;                    // consecutive lanes -> consecutive w
    tl[wv][cin] = src[(size_t)cin * T_ * H_ * W_ + wv];
  }
  __syncthreads();
  unsigned short* dst = xb + (size_t)idx * (W_ * CIN);
  for (int e = tid; e < CIN * W_; e += 256) {
    int cin = e & 31, wv = e >> 5;                    // consecutive lanes -> contiguous stores
    dst[wv * CIN + cin] = f2bf(tl[wv][cin]);
  }
}

// ---------------------------------------------------------------------------
// Kernel 2: fused conv3d (bf16 implicit-GEMM WMMA) + cumsum over T + first-
// spike one-hot + per-t spike counts. One workgroup per (b,h) row; 8 waves =
// 4 N-tiles x 2 M-pairs. Weights live in LDS; x slices stream through a
// 4-slot LDS ring filled by async b128 copies two steps ahead.
// ---------------------------------------------------------------------------
__global__ void __launch_bounds__(256)
conv_spike_kernel(const unsigned short* __restrict__ xb,
                  const unsigned short* __restrict__ wb,
                  const float* __restrict__ bglob, float* __restrict__ out,
                  float* __restrict__ cnt) {
  extern __shared__ __align__(16) char smem[];
  unsigned short* wlds = (unsigned short*)smem;                 // [COUT][KCINP]
  unsigned short* xlds = (unsigned short*)(smem + WLDS_BYTES);  // [NSLOT][3][WPAD][CPAD]

  const int tid   = threadIdx.x;
  const int lane  = tid & 31;
  const int wave  = tid >> 5;
  const int b     = blockIdx.x >> 6;
  const int h     = blockIdx.x & 63;
  const int ntile = wave & 3;      // w0 = ntile*16
  const int mpair = wave >> 2;     // couts mpair*32 .. +31
  const int hi    = lane >> 4;     // K half per ISA fragment layouts
  const int ln    = lane & 15;     // M row / N col in tile

  // ---- one-time LDS init -------------------------------------------------
  // Zero the whole x ring: border columns/rows and OOB t-slices stay zero.
  for (int i = tid; i < (XLDS_BYTES / 16); i += 256)
    ((v4u*)xlds)[i] = (v4u){0u, 0u, 0u, 0u};
  // Async-stage all weights (864 shorts copied into 872-short padded rows).
  for (int e = tid; e < COUT * (KCIN / 8); e += 256) {          // 6912 x b128
    int cout = e / 108, q = e % 108;
    async_g2l_b128(wb + cout * KCIN + q * 8, wlds + cout * KCINP + q * 8);
  }

  // Async-stage one bf16 x slice (rows with valid h only; ring pre-zeroed).
  auto stage_async = [&](int slot, int ts) {
#pragma unroll
    for (int e = tid; e < 3 * 256; e += 256) {                  // 768 x b128
      int hr = e >> 8, rem = e & 255, wv = rem >> 2, chunk = rem & 3;
      int hh = h - 1 + hr;
      char* lp = (char*)(xlds + slot * XSL + (hr * WPAD + (wv + 1)) * CPAD) + chunk * 16;
      if (ts >= 0 && ts < T_) {
        if (hh >= 0 && hh < H_) {
          const unsigned short* gp =
              xb + ((size_t)((b * T_ + ts) * H_ + hh) * W_ + wv) * CIN + chunk * 8;
          async_g2l_b128(gp, lp);
        }
      } else {
        *(v4u*)lp = (v4u){0u, 0u, 0u, 0u};                      // OOB t: re-zero slot
      }
    }
  };

  stage_async(1, 0);            // slot(ts) = (ts+1)%4 ; ts=-1 stays zero (slot 0)
  stage_async(2, 1);
  wait_async0();
  __syncthreads();

  // Per-lane bias columns (C layout: VGPR j -> M=j | M=8+j by lane half).
  float biasv[2][8];
#pragma unroll
  for (int mt = 0; mt < 2; ++mt)
#pragma unroll
    for (int j = 0; j < 8; ++j)
      biasv[mt][j] = bglob[mpair * 32 + mt * 16 + hi * 8 + j];

  v8f memA[2]   = {};           // running membrane (scale folded into weights)
  v8f firedA[2] = {};

  for (int t = 0; t < T_; ++t) {
    stage_async((t + 3) & 3, t + 2);    // fill ts=t+2 while computing t

    v8f acc[2] = {};
#pragma unroll
    for (int kd = 0; kd < 3; ++kd) {
      const int slot = (t + kd) & 3;    // slice ts = t-1+kd
#pragma unroll
      for (int kh = 0; kh < 3; ++kh) {
#pragma unroll
        for (int kw = 0; kw < 3; ++kw) {
          const int tap = (kd * 3 + kh) * 3 + kw;
          // B fragment (32x16): lanes 0-15 K=0..15, lanes 16-31 K=16..31.
          Frag bf_;
          const char* bbp = (const char*)(xlds + slot * XSL +
                              (kh * WPAD + (ntile * 16 + ln + kw)) * CPAD) + hi * 32;
          bf_.q[0] = *(const v4u*)(bbp);
          bf_.q[1] = *(const v4u*)(bbp + 16);
#pragma unroll
          for (int mt = 0; mt < 2; ++mt) {
            // A fragment (16x32): lanes 0-15 K=0..15, lanes 16-31 K=16..31.
            const int cout = mpair * 32 + mt * 16 + ln;
            const char* abp = (const char*)(wlds + cout * KCINP + tap * CIN) + hi * 32;
            Frag af;
            af.q[0] = *(const v4u*)(abp);
            af.q[1] = *(const v4u*)(abp + 16);
            acc[mt] = __builtin_amdgcn_wmma_f32_16x16x32_bf16(
                false, af.bf, false, bf_.bf, (short)0, acc[mt], false, false);
          }
        }
      }
    }

    // Epilogue: membrane cumsum, threshold, first-spike one-hot, store, count.
    float lc = 0.f;
#pragma unroll
    for (int mt = 0; mt < 2; ++mt) {
#pragma unroll
      for (int j = 0; j < 8; ++j) {
        float m = memA[mt][j] + acc[mt][j];
        memA[mt][j] = m;
        float thr = m - biasv[mt][j];
        float spk = (thr > 0.f && firedA[mt][j] == 0.f) ? 1.f : 0.f;
        if (thr > 0.f) firedA[mt][j] = 1.f;
        lc += spk;
        const int cout = mpair * 32 + mt * 16 + hi * 8 + j;
        out[(((size_t)(b * COUT + cout) * T_ + t) * H_ + h) * W_ + ntile * 16 + ln] = spk;
      }
    }
#pragma unroll
    for (int off = 16; off > 0; off >>= 1) lc += __shfl_xor(lc, off, 32);
    if (lane == 0) atomicAdd(&cnt[t], lc);

    wait_async0();              // own async fills done before cross-wave handoff
    __syncthreads();
  }
}

// ---------------------------------------------------------------------------
// Kernel 3: loss = 0.5*total/OUT_N (spk in {0,1}); spread = max_t cnt[t]/(B*COUT*H*W).
// ---------------------------------------------------------------------------
__global__ void finalize_kernel(const float* __restrict__ cnt, float* __restrict__ out) {
  if (threadIdx.x == 0 && blockIdx.x == 0) {
    float total = 0.f, mx = 0.f;
    for (int t = 0; t < T_; ++t) {
      float c = cnt[t];
      total += c;
      mx = (c > mx) ? c : mx;
    }
    out[OUT_N]     = 0.5f * total / (float)OUT_N;
    out[OUT_N + 1] = mx / (float)(B_ * COUT * H_ * W_);
  }
}

extern "C" void kernel_launch(void* const* d_in, const int* in_sizes, int n_in,
                              void* d_out, int out_size, void* d_ws, size_t ws_size,
                              hipStream_t stream) {
  const float* x    = (const float*)d_in[0];   // (4,32,50,64,64)
  const float* w    = (const float*)d_in[1];   // (64,32,3,3,3)
  const float* bias = (const float*)d_in[2];   // (64,)
  // d_in[3] = sigma: only used by the custom-VJP backward; forward ignores it.
  float* out = (float*)d_out;

  unsigned short* wb  = (unsigned short*)d_ws;                       // 110,592 B bf16 weights
  float*          cnt = (float*)((char*)d_ws + 110592);              // 64 floats
  unsigned short* xbt = (unsigned short*)((char*)d_ws + 110848);     // 52,428,800 B bf16 x (transposed)

  prep_kernel<<<1, 256, 0, stream>>>(w, wb, cnt);
  xpose_kernel<<<dim3(B_ * T_ * H_), dim3(256), 0, stream>>>(x, xbt);
  conv_spike_kernel<<<dim3(B_ * H_), dim3(256),
                      WLDS_BYTES + XLDS_BYTES, stream>>>(xbt, wb, bias, out, cnt);
  finalize_kernel<<<1, 64, 0, stream>>>(cnt, out);
}
